// DualTierMiras_6743098655199
// MI455X (gfx1250) — compile-verified, matrix operand order checked by reading, not run on hardware
//
#include <hip/hip_runtime.h>
#include <cstdint>

// ---------------------------------------------------------------------------
// DualTierMiras fused pipeline for gfx1250 (MI455X), bf16 WMMA everywhere.
// B=65536, D=256, H=4, S=64, HD=64.
// Roofline: ~192 MB mandatory HBM traffic (~8.2 us @ 23.3 TB/s); ~39 GFLOP of
// GEMM -> bf16 WMMA keeps compute at/below the memory floor. Intermediates
// (q/out_pre bf16, mix/conf, swizzled weights) live in the 192 MB L2.
// ---------------------------------------------------------------------------

typedef __bf16 bf16_t;
typedef bf16_t v16bf __attribute__((ext_vector_type(16)));
typedef float  v8f   __attribute__((ext_vector_type(8)));
typedef float         vf4 __attribute__((ext_vector_type(4)));
typedef unsigned int  vu4 __attribute__((ext_vector_type(4)));
typedef unsigned int  vu2 __attribute__((ext_vector_type(2)));

union Frag { v16bf v; unsigned int u[8]; unsigned short s[16]; };
union Acc  { v8f v; float f[8]; };

__device__ __forceinline__ unsigned short f2bf(float f) {
  unsigned int u = __float_as_uint(f);
  u += 0x7FFFu + ((u >> 16) & 1u);   // round-to-nearest-even
  return (unsigned short)(u >> 16);
}
__device__ __forceinline__ float bf2f(unsigned short s) {
  return __uint_as_float(((unsigned int)s) << 16);
}
// K index of the low element of u32-pair p within a 16-bit A/B WMMA fragment
// (CDNA5 ISA 7.12.2, 16-bit 16x32 layout).
__device__ __forceinline__ int kpair(int lane, int p) {
  int half = (lane & 16) ? 8 : 0;
  return (p < 4) ? (half + 2 * p) : (16 + half + 2 * (p - 4));
}
__device__ __forceinline__ float sigmoidf_(float x) { return 1.0f / (1.0f + __expf(-x)); }

__device__ __forceinline__ void load_bfrag(Frag& b, const char* base, int byteOfs) {
  vu4 lo = *(const vu4*)(base + byteOfs);
  vu4 hi = *(const vu4*)(base + byteOfs + 16);
  b.u[0] = lo[0]; b.u[1] = lo[1]; b.u[2] = lo[2]; b.u[3] = lo[3];
  b.u[4] = hi[0]; b.u[5] = hi[1]; b.u[6] = hi[2]; b.u[7] = hi[3];
}

// ---------------------------------------------------------------------------
// Prep: swizzle weight W[nout,256] (row-major f32) into bf16 B-fragment layout:
// F[((nt*8+kc)*32+lane)*16 + e] = bf16( W[nt*16+(lane&15), kc*32 + K(lane,e)] )
// ---------------------------------------------------------------------------
__global__ void k_prep_w(const float* __restrict__ W, unsigned short* __restrict__ F,
                         int ntiles) {
  int gid = blockIdx.x * blockDim.x + threadIdx.x;  // gid == (nt*8+kc)*32+lane
  if (gid >= ntiles * 256) return;
  int lane = gid & 31;
  int row = (gid >> 8) * 16 + (lane & 15);
  int kc = (gid >> 5) & 7;
#pragma unroll
  for (int e = 0; e < 16; ++e) {
    int k = kc * 32 + kpair(lane, e >> 1) + (e & 1);
    F[gid * 16 + e] = f2bf(W[row * 256 + k]);
  }
}

__global__ void k_prep_biasgc(const float* __restrict__ bg, const float* __restrict__ bc1,
                              float* __restrict__ bias) {
  int i = blockIdx.x * blockDim.x + threadIdx.x;
  if (i < 256) bias[i] = bg[i];
  else if (i < 384) bias[i] = bc1[i - 256];
}

// knT fragments: B[d,s] = keys[h][s][d]/(||keys[h][s]||+eps).  K-dim = d (64), N = s.
__global__ void k_prep_kn(const float* __restrict__ K, unsigned short* __restrict__ F) {
  int gid = blockIdx.x * blockDim.x + threadIdx.x;  // 1024 = H*4nt*2kc*32
  if (gid >= 1024) return;
  int lane = gid & 31, kc = (gid >> 5) & 1, nt = (gid >> 6) & 3, h = gid >> 8;
  int s = nt * 16 + (lane & 15);
  const float* kr = K + (h * 64 + s) * 64;
  float ss = 0.f;
  for (int d = 0; d < 64; ++d) { float v = kr[d]; ss += v * v; }
  float inv = 1.0f / (sqrtf(ss) + 1e-8f);
#pragma unroll
  for (int e = 0; e < 16; ++e) {
    int d = kc * 32 + kpair(lane, e >> 1) + (e & 1);
    F[gid * 16 + e] = f2bf(kr[d] * inv);
  }
}

// vals fragments: B[s,d] = vals[h][s][d].  K-dim = s (64), N = d.
__global__ void k_prep_val(const float* __restrict__ V, unsigned short* __restrict__ F) {
  int gid = blockIdx.x * blockDim.x + threadIdx.x;
  if (gid >= 1024) return;
  int lane = gid & 31, kc = (gid >> 5) & 1, nt = (gid >> 6) & 3, h = gid >> 8;
  int d = nt * 16 + (lane & 15);
#pragma unroll
  for (int e = 0; e < 16; ++e) {
    int s = kc * 32 + kpair(lane, e >> 1) + (e & 1);
    F[gid * 16 + e] = f2bf(V[(h * 64 + s) * 64 + d]);
  }
}

// ---------------------------------------------------------------------------
// 128-row x 256-col GEMM:  Out = A @ W^T + bias.  8 waves, wave w owns M-tile w.
// A staged in LDS as bf16 (stride 264). B fragments: 16 batched b128-pair loads
// per k-chunk (constant immediate offsets), then a 16-deep WMMA chain.
// Accumulators zero-init; bias added in epilogue (keeps loop regs lean).
// Epilogue staged through LDS (reusing the A buffer) for coalesced b128 stores.
// ---------------------------------------------------------------------------
template <bool ABF16, bool OBF16>
__global__ void k_gemm256(const void* __restrict__ Ain,
                          const unsigned short* __restrict__ BF,
                          const float* __restrict__ bias, void* __restrict__ Out) {
  extern __shared__ __align__(16) char smem[];
  unsigned short* Alds = (unsigned short*)smem;  // [128][264] bf16
  const int tid = threadIdx.x;
  const int rowBase = blockIdx.x * 128;

  if constexpr (ABF16) {
    const unsigned short* A = (const unsigned short*)Ain;
#pragma unroll
    for (int it = 0; it < 8; ++it) {
      int idx = it * 256 + tid;  // 8-ushort unit
      int r = idx >> 5, c8 = (idx & 31) * 8;
      vu4 d = *(const vu4*)(A + (size_t)(rowBase + r) * 256 + c8);
      *(vu4*)(Alds + r * 264 + c8) = d;
    }
  } else {
    const float* A = (const float*)Ain;
#pragma unroll
    for (int it = 0; it < 32; ++it) {
      int idx = it * 256 + tid;  // float4 unit
      int r = idx >> 6, c4 = (idx & 63) * 4;
      vf4 d = __builtin_nontemporal_load((const vf4*)(A + (size_t)(rowBase + r) * 256 + c4));
      vu2 p;
      p[0] = (unsigned int)f2bf(d[0]) | ((unsigned int)f2bf(d[1]) << 16);
      p[1] = (unsigned int)f2bf(d[2]) | ((unsigned int)f2bf(d[3]) << 16);
      *(vu2*)(Alds + r * 264 + c4) = p;
    }
  }
  __syncthreads();

  const int wave = tid >> 5, lane = tid & 31;
  const unsigned short* abase = Alds + (16 * wave + (lane & 15)) * 264;
  const char* bb = (const char*)BF + (size_t)lane * 32;

  Acc acc[16];
#pragma unroll
  for (int nt = 0; nt < 16; ++nt)
#pragma unroll
    for (int i = 0; i < 8; ++i) acc[nt].f[i] = 0.f;

#pragma unroll
  for (int kc = 0; kc < 8; ++kc) {
    Frag a;
#pragma unroll
    for (int p = 0; p < 8; ++p)
      a.u[p] = *(const unsigned int*)(abase + kc * 32 + kpair(lane, p));
    Frag b[16];
#pragma unroll
    for (int nt = 0; nt < 16; ++nt) load_bfrag(b[nt], bb, (nt * 8 + kc) * 1024);
#pragma unroll
    for (int nt = 0; nt < 16; ++nt)
      acc[nt].v = __builtin_amdgcn_wmma_f32_16x16x32_bf16(
          false, a.v, false, b[nt].v, (short)0, acc[nt].v, false, false);
  }

  const int halfsel = (lane >> 4) & 1;
  const float* bptr = bias + (lane & 15);  // + nt*16 as immediate
  __syncthreads();  // A data dead; reuse LDS as store-stage
  if constexpr (OBF16) {
    // stage bf16 (+bias), then 16 coalesced b128 row stores per wave
#pragma unroll
    for (int nt = 0; nt < 16; ++nt) {
      float bv = bptr[nt * 16];
#pragma unroll
      for (int r = 0; r < 8; ++r)
        Alds[(16 * wave + r + 8 * halfsel) * 264 + nt * 16 + (lane & 15)] =
            f2bf(acc[nt].f[r] + bv);
    }
    __syncthreads();
    unsigned short* O = (unsigned short*)Out;
#pragma unroll
    for (int rr = 0; rr < 16; ++rr) {
      vu4 d = *(const vu4*)(Alds + (16 * wave + rr) * 264 + lane * 8);
      *(vu4*)(O + (size_t)(rowBase + 16 * wave + rr) * 256 + lane * 8) = d;
    }
  } else {
    // stage f32 (+bias) in two half-width passes, nontemporal b128 stores
    float* fst = (float*)smem + wave * 2112;  // [16][132] per wave
    float* O = (float*)Out;
#pragma unroll
    for (int ntg = 0; ntg < 2; ++ntg) {
#pragma unroll
      for (int j = 0; j < 8; ++j) {
        float bv = bptr[(ntg * 8 + j) * 16];
#pragma unroll
        for (int r = 0; r < 8; ++r)
          fst[(r + 8 * halfsel) * 132 + j * 16 + (lane & 15)] = acc[ntg * 8 + j].f[r] + bv;
      }
      __syncthreads();
#pragma unroll
      for (int rr = 0; rr < 16; ++rr) {
        vf4 d = *(const vf4*)(fst + rr * 132 + lane * 4);
        __builtin_nontemporal_store(
            d, (vf4*)(O + (size_t)(rowBase + 16 * wave + rr) * 256 + ntg * 128 + lane * 4));
      }
      __syncthreads();
    }
  }
}

// ---------------------------------------------------------------------------
// Gate/conf kernel: y = tanh(context @ [Wg;Wc1]^T + [bg;bc1])  (N=384),
// mix = sigmoid(mix_logit + mean(y[:,0:256])), conf = sigmoid(y[:,256:384]@Wc2+bc2).
// 64 rows/block; waves 0-3 cover nt 0..11, waves 4-7 cover nt 12..23.
// ---------------------------------------------------------------------------
__global__ void k_gate(const float* __restrict__ ctx, const unsigned short* __restrict__ BF,
                       const float* __restrict__ bias, const float* __restrict__ Wc2,
                       const float* __restrict__ bc2, const float* __restrict__ mix_logit,
                       float* __restrict__ mixO, float* __restrict__ confO) {
  extern __shared__ __align__(16) char smem[];
  unsigned short* Alds = (unsigned short*)smem;        // [64][264] bf16
  float* stage = (float*)(smem + 64 * 264 * 2);        // [64][385] f32
  const int tid = threadIdx.x;
  const int rowBase = blockIdx.x * 64;

#pragma unroll
  for (int it = 0; it < 16; ++it) {
    int idx = it * 256 + tid;
    int r = idx >> 6, c4 = (idx & 63) * 4;
    vf4 d = __builtin_nontemporal_load((const vf4*)(ctx + (size_t)(rowBase + r) * 256 + c4));
    vu2 p;
    p[0] = (unsigned int)f2bf(d[0]) | ((unsigned int)f2bf(d[1]) << 16);
    p[1] = (unsigned int)f2bf(d[2]) | ((unsigned int)f2bf(d[3]) << 16);
    *(vu2*)(Alds + r * 264 + c4) = p;
  }
  __syncthreads();

  const int wave = tid >> 5, lane = tid & 31;
  const int m = wave & 3, nh = wave >> 2;
  const unsigned short* abase = Alds + (16 * m + (lane & 15)) * 264;
  const char* bb = (const char*)BF + (size_t)nh * 98304 + (size_t)lane * 32;

  Acc acc[12];
#pragma unroll
  for (int j = 0; j < 12; ++j)
#pragma unroll
    for (int i = 0; i < 8; ++i) acc[j].f[i] = 0.f;

#pragma unroll
  for (int kc = 0; kc < 8; ++kc) {
    Frag a;
#pragma unroll
    for (int p = 0; p < 8; ++p)
      a.u[p] = *(const unsigned int*)(abase + kc * 32 + kpair(lane, p));
    Frag b[12];
#pragma unroll
    for (int j = 0; j < 12; ++j) load_bfrag(b[j], bb, (j * 8 + kc) * 1024);
#pragma unroll
    for (int j = 0; j < 12; ++j)
      acc[j].v = __builtin_amdgcn_wmma_f32_16x16x32_bf16(
          false, a.v, false, b[j].v, (short)0, acc[j].v, false, false);
  }
  const int halfsel = (lane >> 4) & 1;
  const float* bptr = bias + nh * 192 + (lane & 15);  // + j*16 as immediate
#pragma unroll
  for (int j = 0; j < 12; ++j) {
    float bv = bptr[j * 16];
    int col = (nh * 12 + j) * 16 + (lane & 15);
#pragma unroll
    for (int r = 0; r < 8; ++r)
      stage[(16 * m + r + 8 * halfsel) * 385 + col] = tanhf(acc[j].f[r] + bv);
  }
  __syncthreads();

  if (tid < 64) {
    const float* sr = stage + tid * 385;
    float gs = 0.f;
    for (int c = 0; c < 256; ++c) gs += sr[c];
    float mix = sigmoidf_(mix_logit[0] + gs * (1.0f / 256.0f));
    float cd = bc2[0];
    for (int c = 0; c < 128; ++c) cd += sr[256 + c] * Wc2[c];
    mixO[rowBase + tid] = mix;
    confO[rowBase + tid] = sigmoidf_(cd);
  }
}

// ---------------------------------------------------------------------------
// Attention: per 64-row block. qn@knT (WMMA) -> in-register softmax ->
// P@vals (WMMA) per tier -> combine with mix/conf -> write out_pre bf16
// in-place over the q buffer (block-local rows only).
// Wave w: M-tile = w&3, heads {2*(w>>2), 2*(w>>2)+1}.
// ---------------------------------------------------------------------------
__global__ void k_attn(unsigned short* __restrict__ qbuf,
                       const unsigned short* __restrict__ knFf,
                       const unsigned short* __restrict__ knFd,
                       const unsigned short* __restrict__ vlFf,
                       const unsigned short* __restrict__ vlFd,
                       const float* __restrict__ mixO, const float* __restrict__ confO) {
  extern __shared__ __align__(16) char smem[];
  unsigned short* qA = (unsigned short*)smem;                       // [64][264] bf16
  float* invn = (float*)(smem + 64 * 264 * 2);                      // [64][4]
  unsigned short* pbuf = (unsigned short*)(smem + 64 * 264 * 2 + 64 * 4 * 4);  // 8x[16][66]
  const int tid = threadIdx.x;
  const int rowBase = blockIdx.x * 64;

#pragma unroll
  for (int it = 0; it < 4; ++it) {
    int idx = it * 256 + tid;
    int r = idx >> 5, c8 = (idx & 31) * 8;
    *(vu4*)(qA + r * 264 + c8) =
        *(const vu4*)(qbuf + (size_t)(rowBase + r) * 256 + c8);
  }
  __syncthreads();

  {  // per-(row,head) inverse norms: 256 threads <-> 64 rows x 4 heads
    int r = tid >> 2, h = tid & 3;
    float ss = 0.f;
    for (int d = 0; d < 64; ++d) {
      float v = bf2f(qA[r * 264 + h * 64 + d]);
      ss += v * v;
    }
    invn[r * 4 + h] = 1.0f / (sqrtf(ss) + 1e-8f);
  }
  __syncthreads();

  const int wave = tid >> 5, lane = tid & 31;
  const int m = wave & 3, hbase = (wave >> 2) * 2;
  const int halfsel = (lane >> 4) & 1;
  unsigned short* pw = pbuf + wave * (16 * 66);

  float mixv[8], confv[8];
  int grow[8];
#pragma unroll
  for (int v = 0; v < 8; ++v) {
    grow[v] = rowBase + 16 * m + v + 8 * halfsel;
    mixv[v] = mixO[grow[v]];
    confv[v] = confO[grow[v]];
  }

  for (int hh = 0; hh < 2; ++hh) {
    const int h = hbase + hh;
    const float invA = invn[(16 * m + (lane & 15)) * 4 + h];
    Acc vout[2][4];

    for (int tier = 0; tier < 2; ++tier) {
      const char* knb = (const char*)(tier ? knFd : knFf) + (size_t)h * 8192 + (size_t)lane * 32;
      const char* vlb = (const char*)(tier ? vlFd : vlFf) + (size_t)h * 8192 + (size_t)lane * 32;

      Acc sim[4];
#pragma unroll
      for (int nt = 0; nt < 4; ++nt)
#pragma unroll
        for (int i = 0; i < 8; ++i) sim[nt].f[i] = 0.f;

#pragma unroll
      for (int kc = 0; kc < 2; ++kc) {
        Frag b[4];
#pragma unroll
        for (int nt = 0; nt < 4; ++nt) load_bfrag(b[nt], knb, (nt * 2 + kc) * 1024);
        Frag a;  // qn fragment, scaled on the fly by the row/head inv-norm
#pragma unroll
        for (int p = 0; p < 8; ++p) {
          unsigned int u = *(const unsigned int*)(
              qA + (16 * m + (lane & 15)) * 264 + h * 64 + kc * 32 + kpair(lane, p));
          float lo = bf2f((unsigned short)(u & 0xFFFFu)) * invA;
          float hi = bf2f((unsigned short)(u >> 16)) * invA;
          a.u[p] = (unsigned int)f2bf(lo) | ((unsigned int)f2bf(hi) << 16);
        }
#pragma unroll
        for (int nt = 0; nt < 4; ++nt)
          sim[nt].v = __builtin_amdgcn_wmma_f32_16x16x32_bf16(
              false, a.v, false, b[nt].v, (short)0, sim[nt].v, false, false);
      }

      // softmax per row: row's 64 sims live in 4 tiles x 16 lanes of one half
#pragma unroll
      for (int v = 0; v < 8; ++v) {
        float mx = sim[0].f[v];
#pragma unroll
        for (int nt = 1; nt < 4; ++nt) mx = fmaxf(mx, sim[nt].f[v]);
        mx = fmaxf(mx, __shfl_xor(mx, 1, 32));
        mx = fmaxf(mx, __shfl_xor(mx, 2, 32));
        mx = fmaxf(mx, __shfl_xor(mx, 4, 32));
        mx = fmaxf(mx, __shfl_xor(mx, 8, 32));
        float sum = 0.f;
#pragma unroll
        for (int nt = 0; nt < 4; ++nt) {
          float e = __expf(sim[nt].f[v] - mx);
          sim[nt].f[v] = e;
          sum += e;
        }
        sum += __shfl_xor(sum, 1, 32);
        sum += __shfl_xor(sum, 2, 32);
        sum += __shfl_xor(sum, 4, 32);
        sum += __shfl_xor(sum, 8, 32);
        float rs = 1.0f / sum;
#pragma unroll
        for (int nt = 0; nt < 4; ++nt) sim[nt].f[v] *= rs;
      }

      // stage P (C layout -> row-major bf16 in wave-private LDS)
#pragma unroll
      for (int nt = 0; nt < 4; ++nt)
#pragma unroll
        for (int v = 0; v < 8; ++v)
          pw[(v + 8 * halfsel) * 66 + nt * 16 + (lane & 15)] = f2bf(sim[nt].f[v]);
      __syncthreads();

      // out_tier = P @ vals
      Acc* oa = vout[tier];
#pragma unroll
      for (int nt = 0; nt < 4; ++nt)
#pragma unroll
        for (int i = 0; i < 8; ++i) oa[nt].f[i] = 0.f;
#pragma unroll
      for (int kc = 0; kc < 2; ++kc) {
        Frag b[4];
#pragma unroll
        for (int nt = 0; nt < 4; ++nt) load_bfrag(b[nt], vlb, (nt * 2 + kc) * 1024);
        Frag a;
#pragma unroll
        for (int p = 0; p < 8; ++p)
          a.u[p] = *(const unsigned int*)(pw + (lane & 15) * 66 + kc * 32 + kpair(lane, p));
#pragma unroll
        for (int nt = 0; nt < 4; ++nt)
          oa[nt].v = __builtin_amdgcn_wmma_f32_16x16x32_bf16(
              false, a.v, false, b[nt].v, (short)0, oa[nt].v, false, false);
      }
      __syncthreads();  // WAR: pbuf reused next tier/head
    }

    // combine tiers, apply conf, write out_pre bf16 (in place over q buffer)
#pragma unroll
    for (int nt = 0; nt < 4; ++nt)
#pragma unroll
      for (int v = 0; v < 8; ++v) {
        float f = confv[v] * (mixv[v] * vout[0][nt].f[v] +
                              (1.0f - mixv[v]) * vout[1][nt].f[v]);
        qbuf[(size_t)grow[v] * 256 + h * 64 + nt * 16 + (lane & 15)] = f2bf(f);
      }
  }
}

// ---------------------------------------------------------------------------
extern "C" void kernel_launch(void* const* d_in, const int* in_sizes, int n_in,
                              void* d_out, int out_size, void* d_ws, size_t ws_size,
                              hipStream_t stream) {
  const float* query   = (const float*)d_in[0];
  const float* context = (const float*)d_in[1];
  const float* fkeys   = (const float*)d_in[2];
  const float* fvals   = (const float*)d_in[3];
  const float* dkeys   = (const float*)d_in[4];
  const float* dvals   = (const float*)d_in[5];
  const float* Wq  = (const float*)d_in[6];
  const float* bq  = (const float*)d_in[7];
  const float* Wg  = (const float*)d_in[8];
  const float* bg  = (const float*)d_in[9];
  const float* Wc1 = (const float*)d_in[10];
  const float* bc1 = (const float*)d_in[11];
  const float* Wc2 = (const float*)d_in[12];
  const float* bc2 = (const float*)d_in[13];
  const float* Wo  = (const float*)d_in[14];
  const float* bo  = (const float*)d_in[15];
  const float* mix_logit = (const float*)d_in[18];
  float* out = (float*)d_out;

  char* ws = (char*)d_ws;
  unsigned short* WqF  = (unsigned short*)(ws + 0);         // 128 KB
  unsigned short* WgcF = (unsigned short*)(ws + 131072);    // 192 KB (Wg 16 tiles + Wc1 8 tiles)
  unsigned short* WoF  = (unsigned short*)(ws + 327680);    // 128 KB
  float*          biasgc = (float*)(ws + 458752);           // 384 f32
  unsigned short* knFf = (unsigned short*)(ws + 460288);    // 32 KB each
  unsigned short* knFd = (unsigned short*)(ws + 493056);
  unsigned short* vlFf = (unsigned short*)(ws + 525824);
  unsigned short* vlFd = (unsigned short*)(ws + 558592);
  float*          mixO = (float*)(ws + 591360);             // 256 KB
  float*          confO = (float*)(ws + 853504);            // 256 KB
  unsigned short* qbuf = (unsigned short*)(ws + 1115648);   // 32 MB (q, then out_pre)

  // --- prep (tiny, L2-resident) ---
  k_prep_w<<<16, 256, 0, stream>>>(Wq, WqF, 16);
  k_prep_w<<<16, 256, 0, stream>>>(Wg, WgcF, 16);
  k_prep_w<<<8, 256, 0, stream>>>(Wc1, WgcF + 16 * 4096, 8);
  k_prep_w<<<16, 256, 0, stream>>>(Wo, WoF, 16);
  k_prep_biasgc<<<2, 256, 0, stream>>>(bg, bc1, biasgc);
  k_prep_kn<<<4, 256, 0, stream>>>(fkeys, knFf);
  k_prep_kn<<<4, 256, 0, stream>>>(dkeys, knFd);
  k_prep_val<<<4, 256, 0, stream>>>(fvals, vlFf);
  k_prep_val<<<4, 256, 0, stream>>>(dvals, vlFd);

  // --- q = query @ Wq^T + bq  (bf16 out) ---
  k_gemm256<false, true><<<512, 256, 128 * 264 * 2, stream>>>(query, WqF, bq, qbuf);

  // --- mix / conf from context ---
  k_gate<<<1024, 256, 64 * 264 * 2 + 64 * 385 * 4, stream>>>(
      context, WgcF, biasgc, Wc2, bc2, mix_logit, mixO, confO);

  // --- dual-tier cosine attention + gating -> out_pre (in place) ---
  k_attn<<<1024, 256, 64 * 264 * 2 + 64 * 4 * 4 + 8 * 16 * 66 * 2, stream>>>(
      qbuf, knFf, knFd, vlFf, vlFd, mixO, confO);

  // --- out = out_pre @ Wo^T + bo  (f32) ---
  k_gemm256<true, false><<<512, 256, 128 * 264 * 2, stream>>>(qbuf, WoF, bo, out);
}
